// HeatMapRegressor_30013231464923
// MI455X (gfx1250) — matrix operand order repaired and checked by reading, not compiled, not run on hardware
//
#include <hip/hip_runtime.h>
#include <hip/hip_bf16.h>
#include <stdint.h>

typedef __attribute__((ext_vector_type(16))) __bf16 v16bf;
typedef __attribute__((ext_vector_type(8)))  __bf16 v8bf;
typedef __attribute__((ext_vector_type(8)))  float  v8f;

// ---------------------------------------------------------------------------
// Degree: deg[dst] += 1  (graph-invariant, computed once)
// ---------------------------------------------------------------------------
__global__ void sage_degree(const int* __restrict__ dst, float* __restrict__ deg, int E) {
    for (long long t = (long long)blockIdx.x * blockDim.x + threadIdx.x; t < E;
         t += (long long)gridDim.x * blockDim.x) {
        atomicAdd(&deg[dst[(int)t]], 1.0f);
    }
}

// ---------------------------------------------------------------------------
// Edge aggregation: agg[dst, f] += h[src, f]   (f32 accumulation)
// T = float (layer 0 / pos) or __bf16 (layers 1,2: halves the gather bytes;
// the feature table lives in the 192MB L2, so this is the hot loop).
// ---------------------------------------------------------------------------
template <typename T, int DIN>
__global__ void sage_aggregate(const int* __restrict__ src, const int* __restrict__ dst,
                               const T* __restrict__ h, float* __restrict__ agg, int E) {
    long long total = (long long)E * DIN;
    for (long long t = (long long)blockIdx.x * blockDim.x + threadIdx.x; t < total;
         t += (long long)gridDim.x * blockDim.x) {
        int e = (int)(t / DIN);
        int f = (int)(t - (long long)e * DIN);
        long long s = src[e];
        long long d = dst[e];
        atomicAdd(&agg[d * DIN + f], (float)h[s * DIN + f]);
    }
}

// ---------------------------------------------------------------------------
// Layer 0: din=3 -> dout=64, plain VALU (K too small for WMMA). Output bf16.
// ---------------------------------------------------------------------------
__global__ void sage_layer0(const float* __restrict__ pos, const float* __restrict__ agg,
                            const float* __restrict__ deg, const float* __restrict__ Wl,
                            const float* __restrict__ bl, const float* __restrict__ Wr,
                            __bf16* __restrict__ out, int N) {
    int t = blockIdx.x * blockDim.x + threadIdx.x;
    if (t >= N * 64) return;
    int n = t >> 6;
    int o = t & 63;
    float invd = 1.0f / fmaxf(deg[n], 1.0f);
    float v = bl[o];
#pragma unroll
    for (int k = 0; k < 3; ++k) {
        v += agg[n * 3 + k] * invd * Wl[k * 64 + o] + pos[n * 3 + k] * Wr[k * 64 + o];
    }
    v = fmaxf(v, 0.0f);  // ReLU after layer 0
    out[t] = (__bf16)v;
}

// ---------------------------------------------------------------------------
// Layers 1,2: fused SAGE GEMM via V_WMMA_F32_16X16X32_BF16.
//   out[n,:] = [mean(n) | x(n)] (1 x 2*DIN)  @  [Wl ; Wr] (2*DIN x DOUT) + b
// One wave computes a 16-node tile; 8 waves/block share weights in LDS.
// Weights are staged PRE-SWIZZLED into the WMMA B-fragment layout so each
// lane's fragment is 32 contiguous LDS bytes (2x ds_load_b128, no u16 spam).
// DOUTP = DOUT padded to a multiple of 16 (zero-padded columns).
// ---------------------------------------------------------------------------
template <int DIN, int DOUT, int DOUTP, bool RELU, bool OUTF32>
__global__ void sage_gemm_wmma(const __bf16* __restrict__ hprev, const float* __restrict__ agg,
                               const float* __restrict__ deg, const float* __restrict__ Wl,
                               const float* __restrict__ bl, const float* __restrict__ Wr,
                               void* __restrict__ outv, int N) {
    constexpr int K = 2 * DIN;       // 128
    constexpr int KSTEPS = K / 32;   // 4 WMMA k-steps
    constexpr int NFRAG = DOUTP / 16;
    __shared__ alignas(32) __bf16 Bswz[NFRAG * KSTEPS * 32 * 16];
    __shared__ float bias[DOUTP];

    // Stage weights swizzled into fragment layout:
    //   Bswz[((ot*KSTEPS+s)*32 + lane)*16 + e] = W[kg][col]
    //   kg = s*32 + 16*(e/8) + (lane/16)*8 + (e%8),  col = ot*16 + (lane%16)
    for (int i = threadIdx.x; i < NFRAG * KSTEPS * 32 * 16; i += blockDim.x) {
        int e  = i & 15;
        int ln = (i >> 4) & 31;
        int fs = i >> 9;             // ot*KSTEPS + s
        int s  = fs % KSTEPS;
        int ot = fs / KSTEPS;
        int kg = s * 32 + (e >> 3) * 16 + (ln >> 4) * 8 + (e & 7);
        int c  = ot * 16 + (ln & 15);
        float w = 0.0f;
        if (c < DOUT) w = (kg < DIN) ? Wl[kg * DOUT + c] : Wr[(kg - DIN) * DOUT + c];
        Bswz[i] = (__bf16)w;
    }
    for (int c = threadIdx.x; c < DOUTP; c += blockDim.x) bias[c] = (c < DOUT) ? bl[c] : 0.0f;
    __syncthreads();

    const int wave = threadIdx.x >> 5;
    const int lane = threadIdx.x & 31;
    const int tile = blockIdx.x * (blockDim.x >> 5) + wave;
    if (tile * 16 >= N) return;  // wave-uniform: EXEC stays all-ones for WMMA

    const int mrow = lane & 15;  // M / N position within tile
    const int hi   = lane >> 4;  // lane half selects K sub-range
    const long long node = (long long)tile * 16 + mrow;
    const float invd = 1.0f / fmaxf(deg[node], 1.0f);

    // Build A fragments once per node tile (reused for every output tile).
    // 16-bit A 16x32 layout: lane half hi, element e -> K = 16*(e/8) + 8*hi + (e%8)
    //  => per lane, elements 0..7 and 8..15 are two contiguous 8-feature runs.
    v16bf A[KSTEPS];
#pragma unroll
    for (int s = 0; s < KSTEPS; ++s) {
        const int kbase = s * 32 + hi * 8;  // run0 start; run1 = +16
        if (kbase + 16 + 7 < DIN) {
            // Mean path: vectorized f32 loads, scale by 1/deg, convert to bf16.
            const float* arow = agg + node * DIN + kbase;
            float4 a0 = *(const float4*)(arow + 0);
            float4 a1 = *(const float4*)(arow + 4);
            float4 a2 = *(const float4*)(arow + 16);
            float4 a3 = *(const float4*)(arow + 20);
            A[s][0]  = (__bf16)(a0.x * invd); A[s][1]  = (__bf16)(a0.y * invd);
            A[s][2]  = (__bf16)(a0.z * invd); A[s][3]  = (__bf16)(a0.w * invd);
            A[s][4]  = (__bf16)(a1.x * invd); A[s][5]  = (__bf16)(a1.y * invd);
            A[s][6]  = (__bf16)(a1.z * invd); A[s][7]  = (__bf16)(a1.w * invd);
            A[s][8]  = (__bf16)(a2.x * invd); A[s][9]  = (__bf16)(a2.y * invd);
            A[s][10] = (__bf16)(a2.z * invd); A[s][11] = (__bf16)(a2.w * invd);
            A[s][12] = (__bf16)(a3.x * invd); A[s][13] = (__bf16)(a3.y * invd);
            A[s][14] = (__bf16)(a3.z * invd); A[s][15] = (__bf16)(a3.w * invd);
        } else {
            // Root path: bf16 already -- pure 16B vector loads, register repack.
            const __bf16* hrow = hprev + node * DIN + (kbase - DIN);
            v8bf h0 = *(const v8bf*)(hrow + 0);
            v8bf h1 = *(const v8bf*)(hrow + 16);
#pragma unroll
            for (int e = 0; e < 8; ++e) {
                A[s][e]     = h0[e];
                A[s][e + 8] = h1[e];
            }
        }
    }

#pragma unroll
    for (int ot = 0; ot < NFRAG; ++ot) {
        v8f C = {};
#pragma unroll
        for (int s = 0; s < KSTEPS; ++s) {
            // Swizzled fragment: 32 contiguous bytes per lane.
            v16bf B = *(const v16bf*)&Bswz[((ot * KSTEPS + s) * 32 + lane) * 16];
            C = __builtin_amdgcn_wmma_f32_16x16x32_bf16(false, A[s], false, B, (short)0, C,
                                                        false, false);
        }
        // Epilogue: C/D layout lane half hi, VGPR r -> M = r + 8*hi, N = mrow.
#pragma unroll
        for (int r = 0; r < 8; ++r) {
            int m = r + hi * 8;
            long long onode = (long long)tile * 16 + m;
            int col = ot * 16 + mrow;
            float v = C[r] + bias[col];
            if (RELU) v = fmaxf(v, 0.0f);
            if (col < DOUT) {
                if (OUTF32) ((float*)outv)[onode * DOUT + col] = v;
                else        ((__bf16*)outv)[onode * DOUT + col] = (__bf16)v;
            }
        }
    }
}

// ---------------------------------------------------------------------------
extern "C" void kernel_launch(void* const* d_in, const int* in_sizes, int n_in,
                              void* d_out, int out_size, void* d_ws, size_t ws_size,
                              hipStream_t stream) {
    const float* pos = (const float*)d_in[0];
    const int*   ei  = (const int*)d_in[1];
    const float* Wl0 = (const float*)d_in[2];
    const float* bl0 = (const float*)d_in[3];
    const float* Wr0 = (const float*)d_in[4];
    const float* Wl1 = (const float*)d_in[5];
    const float* bl1 = (const float*)d_in[6];
    const float* Wr1 = (const float*)d_in[7];
    const float* Wl2 = (const float*)d_in[8];
    const float* bl2 = (const float*)d_in[9];
    const float* Wr2 = (const float*)d_in[10];

    const int N = in_sizes[0] / 3;
    const int E = in_sizes[1] / 2;
    const int* src = ei;
    const int* dst = ei + E;

    // Workspace carve-out (256B aligned).
    size_t off = 0;
    auto carve = [&](size_t bytes) -> void* {
        void* p = (char*)d_ws + off;
        off = (off + bytes + 255) & ~(size_t)255;
        return p;
    };
    float*  deg = (float*)carve((size_t)N * 4);
    float*  agg = (float*)carve((size_t)N * 64 * 4);
    __bf16* h1  = (__bf16*)carve((size_t)N * 64 * 2);
    __bf16* h2  = (__bf16*)carve((size_t)N * 64 * 2);
    (void)ws_size; (void)n_in; (void)out_size;

    const int TB = 256;
    auto gridFor = [&](long long total) -> int {
        long long b = (total + TB - 1) / TB;
        if (b > (1ll << 22)) b = (1ll << 22);  // grid-stride handles the rest
        return (int)b;
    };

    // Degrees (reused by all layers).
    hipMemsetAsync(deg, 0, (size_t)N * 4, stream);
    sage_degree<<<gridFor(E), TB, 0, stream>>>(dst, deg, E);

    // ---- Layer 0: pos(f32,3) -> h1(bf16,64), ReLU ----
    hipMemsetAsync(agg, 0, (size_t)N * 3 * 4, stream);
    sage_aggregate<float, 3><<<gridFor((long long)E * 3), TB, 0, stream>>>(src, dst, pos, agg, E);
    sage_layer0<<<((long long)N * 64 + TB - 1) / TB, TB, 0, stream>>>(pos, agg, deg, Wl0, bl0, Wr0,
                                                                      h1, N);

    const int tiles  = (N + 15) / 16;
    const int blocks = (tiles + 7) / 8;  // 8 waves (16-node tiles) per block

    // ---- Layer 1: h1 -> h2 (bf16,64), ReLU, WMMA ----
    hipMemsetAsync(agg, 0, (size_t)N * 64 * 4, stream);
    sage_aggregate<__bf16, 64><<<gridFor((long long)E * 64), TB, 0, stream>>>(src, dst, h1, agg, E);
    sage_gemm_wmma<64, 64, 64, true, false><<<blocks, TB, 0, stream>>>(h1, agg, deg, Wl1, bl1, Wr1,
                                                                       (void*)h2, N);

    // ---- Layer 2: h2 -> out (f32,68), no activation, WMMA (dout padded to 80) ----
    hipMemsetAsync(agg, 0, (size_t)N * 64 * 4, stream);
    sage_aggregate<__bf16, 64><<<gridFor((long long)E * 64), TB, 0, stream>>>(src, dst, h2, agg, E);
    sage_gemm_wmma<64, 68, 80, false, true><<<blocks, TB, 0, stream>>>(h2, agg, deg, Wl2, bl2, Wr2,
                                                                       d_out, N);
}
